// _GNN_S2_37623913513356
// MI455X (gfx1250) — matrix operand order repaired and checked by reading, not compiled, not run on hardware
//
#include <hip/hip_runtime.h>

#define Nn 27
#define NNE 729
#define Cc 512
#define Bb 64
#define BN_ROWS (Bb * Nn)      // 1728
#define BE_ROWS (Bb * NNE)     // 46656
#define XSZ (BN_ROWS * Cc)     // 884736
#define ESZ (BE_ROWS * Cc)     // 23887872
#define CC2 (Cc * Cc)          // 262144
#define BN_EPS 1e-5f
#define NTG (Cc / 64)          // 8 column-groups of 64

typedef __attribute__((ext_vector_type(16))) __bf16 v16bf;
typedef __attribute__((ext_vector_type(8)))  float  v8f;

__device__ __forceinline__ void splitf(float x, __bf16& h, __bf16& l) {
  __bf16 hh = (__bf16)x;
  h = hh;
  l = (__bf16)(x - (float)hh);
}

// 16x64 output strip per wave: 4 accumulators share one A fragment.
// OUT[r][n] = sum_k A[r][k] * W[n][k]; bf16 hi/lo split => near-fp32.
// WhBase/WlBase point at W row (ncol_base + ln); tiles are 16 rows apart.
__device__ __forceinline__ void gemm_tile4(const float* __restrict__ Arow,
                                           const __bf16* __restrict__ WhBase,
                                           const __bf16* __restrict__ WlBase,
                                           int sub, v8f acc[4]) {
#pragma unroll 4
  for (int kb = 0; kb < Cc; kb += 32) {
    v16bf ah, al;
    const float* s0 = Arow + kb + 16 * sub;       // A elems 0..7  : K = kb+16*sub+e
    const float* s1 = Arow + kb + 16 * sub + 8;   // A elems 8..15 : K = kb+16*sub+8+e
#pragma unroll
    for (int e = 0; e < 8; ++e) { __bf16 h, l; splitf(s0[e], h, l); ah[e] = h; al[e] = l; }
#pragma unroll
    for (int e = 0; e < 8; ++e) { __bf16 h, l; splitf(s1[e], h, l); ah[8 + e] = h; al[8 + e] = l; }
#pragma unroll
    for (int t = 0; t < 4; ++t) {
      // B: lane group sub covers K = kb + 16*sub .. +15 contiguously (32B aligned)
      v16bf bh = *(const v16bf*)(WhBase + (size_t)t * 16 * Cc + kb + 16 * sub);
      v16bf bl = *(const v16bf*)(WlBase + (size_t)t * 16 * Cc + kb + 16 * sub);
      acc[t] = __builtin_amdgcn_wmma_f32_16x16x32_bf16(false, ah, false, bh, (short)0, acc[t], false, false);
      acc[t] = __builtin_amdgcn_wmma_f32_16x16x32_bf16(false, ah, false, bl, (short)0, acc[t], false, false);
      acc[t] = __builtin_amdgcn_wmma_f32_16x16x32_bf16(false, al, false, bh, (short)0, acc[t], false, false);
    }
  }
}

// ---- weight pre-split: fp32 -> bf16 hi/lo planes -------------------------
__global__ __launch_bounds__(256) void k_gnn_wsplit(const float* __restrict__ W,
                                                    __bf16* __restrict__ hi,
                                                    __bf16* __restrict__ lo, int n) {
  int i = blockIdx.x * blockDim.x + threadIdx.x;
  if (i < n) { __bf16 h, l; splitf(W[i], h, l); hi[i] = h; lo[i] = l; }
}

// ---- x projections: xu=x@Wu^T, xv=x@Wv^T, xa=x@Wa^T, xb=x@Wb^T -----------
__global__ __launch_bounds__(256) void k_gnn_gemm_x(const float* __restrict__ X,
                                                    const __bf16* __restrict__ whi,
                                                    const __bf16* __restrict__ wlo,
                                                    float* __restrict__ xu, float* __restrict__ xv,
                                                    float* __restrict__ xa, float* __restrict__ xb) {
  const int RT = BN_ROWS / 16;  // 108
  int wid = (blockIdx.x * blockDim.x + threadIdx.x) >> 5;
  int lane = threadIdx.x & 31;
  if (wid >= 4 * RT * NTG) return;
  int w = wid / (RT * NTG);
  int rem = wid - w * (RT * NTG);
  int rt = rem / NTG, ntg = rem - rt * NTG;
  int sub = lane >> 4, ln = lane & 15;
  int r = rt * 16 + ln;
  size_t nrow = (size_t)(ntg * 64 + ln) * Cc;
  const __bf16* Wh = whi + (size_t)w * CC2 + nrow;
  const __bf16* Wl = wlo + (size_t)w * CC2 + nrow;
  v8f acc[4] = {{0.f,0.f,0.f,0.f,0.f,0.f,0.f,0.f}, {0.f,0.f,0.f,0.f,0.f,0.f,0.f,0.f},
                {0.f,0.f,0.f,0.f,0.f,0.f,0.f,0.f}, {0.f,0.f,0.f,0.f,0.f,0.f,0.f,0.f}};
  gemm_tile4(X + (size_t)r * Cc, Wh, Wl, sub, acc);
  float* outs[4] = {xu, xv, xa, xb};
  float* O = outs[w];
  int row0 = rt * 16 + 8 * sub;
#pragma unroll
  for (int v = 0; v < 8; ++v) {
    size_t rbase = (size_t)(row0 + v) * Cc;
#pragma unroll
    for (int t = 0; t < 4; ++t) O[rbase + ntg * 64 + t * 16 + ln] = acc[t][v];
  }
}

// ---- edge GEMM + fused broadcast adds: m = edge@We^T + Ax(rep) + Bx(tile) -
__global__ __launch_bounds__(256) void k_gnn_gemm_edge(const float* __restrict__ E,
                                                       const __bf16* __restrict__ Wh5,
                                                       const __bf16* __restrict__ Wl5,
                                                       const float* __restrict__ xa,
                                                       const float* __restrict__ xb,
                                                       float* __restrict__ m) {
  const int RT = BE_ROWS / 16;  // 2916
  int wid = (blockIdx.x * blockDim.x + threadIdx.x) >> 5;
  int lane = threadIdx.x & 31;
  if (wid >= RT * NTG) return;
  int rt = wid / NTG, ntg = wid - rt * NTG;
  int sub = lane >> 4, ln = lane & 15;
  int r = rt * 16 + ln;
  size_t nrow = (size_t)(ntg * 64 + ln) * Cc;
  v8f acc[4] = {{0.f,0.f,0.f,0.f,0.f,0.f,0.f,0.f}, {0.f,0.f,0.f,0.f,0.f,0.f,0.f,0.f},
                {0.f,0.f,0.f,0.f,0.f,0.f,0.f,0.f}, {0.f,0.f,0.f,0.f,0.f,0.f,0.f,0.f}};
  gemm_tile4(E + (size_t)r * Cc, Wh5 + nrow, Wl5 + nrow, sub, acc);
#pragma unroll
  for (int v = 0; v < 8; ++v) {
    int row = rt * 16 + 8 * sub + v;
    int b = row / NNE;
    int ij = row - b * NNE;
    int i = ij / Nn;
    int j = ij - i * Nn;
    const float* xai = xa + ((size_t)(b * Nn + i)) * Cc;
    const float* xbj = xb + ((size_t)(b * Nn + j)) * Cc;
    size_t rbase = (size_t)row * Cc;
#pragma unroll
    for (int t = 0; t < 4; ++t) {
      int col = ntg * 64 + t * 16 + ln;
      m[rbase + col] = acc[t][v] + xai[col] + xbj[col];
    }
  }
}

// ---- BatchNorm stats: one block per channel, reduce over (B, C) ----------
__global__ __launch_bounds__(256) void k_gnn_bnstats(const float* __restrict__ t, int nch,
                                                     float* __restrict__ mean, float* __restrict__ var) {
  int ch = blockIdx.x;
  int tid = threadIdx.x;
  float s = 0.f, s2 = 0.f;
  for (int e = tid; e < Bb * Cc; e += 256) {
    int b = e >> 9, c = e & 511;
    float v = t[((size_t)b * nch + ch) * Cc + c];
    s += v;
    s2 += v * v;
  }
  __shared__ float sh[256], sh2[256];
  sh[tid] = s; sh2[tid] = s2;
  __syncthreads();
  for (int o = 128; o > 0; o >>= 1) {
    if (tid < o) { sh[tid] += sh[tid + o]; sh2[tid] += sh2[tid + o]; }
    __syncthreads();
  }
  if (tid == 0) {
    const float inv = 1.f / (float)(Bb * Cc);
    float mu = sh[0] * inv;
    mean[ch] = mu;
    var[ch] = sh2[0] * inv - mu * mu;   // biased variance, matches reference
  }
}

// ---- edge residual (float4): eOut = eIn + relu(bn(m)) --------------------
__global__ __launch_bounds__(256) void k_gnn_edge_update(const float4* __restrict__ eIn,
                                                         const float4* __restrict__ m,
                                                         const float* __restrict__ mean,
                                                         const float* __restrict__ var,
                                                         const float* __restrict__ gw,
                                                         const float* __restrict__ gb,
                                                         float4* __restrict__ eOut) {
  size_t idx = (size_t)blockIdx.x * 256 + threadIdx.x;
  if (idx >= (size_t)(ESZ / 4)) return;
  int ch = (int)((idx >> 7) % NNE);               // 128 float4 per 512-wide row
  float scale = rsqrtf(var[ch] + BN_EPS) * gw[ch];
  float mu = mean[ch], bias = gb[ch];
  float4 v = m[idx];
  float4 e = eIn[idx];
  float4 o;
  o.x = e.x + fmaxf((v.x - mu) * scale + bias, 0.f);
  o.y = e.y + fmaxf((v.y - mu) * scale + bias, 0.f);
  o.z = e.z + fmaxf((v.z - mu) * scale + bias, 0.f);
  o.w = e.w + fmaxf((v.w - mu) * scale + bias, 0.f);
  eOut[idx] = o;
}

// ---- sigmoid -> softmax over j -> weighted aggregate -> +xu --------------
__global__ __launch_bounds__(256) void k_gnn_softmax_agg(const float* __restrict__ eNew,
                                                         const float* __restrict__ xv,
                                                         const float* __restrict__ xu,
                                                         float* __restrict__ xtmp) {
  int idx = blockIdx.x * 256 + threadIdx.x;
  if (idx >= XSZ) return;
  int c = idx & 511;
  int bi = idx >> 9;          // b*27 + i
  int b = bi / Nn;
  int i = bi - b * Nn;
  const float* ebase = eNew + (((size_t)b * NNE + (size_t)i * Nn) << 9) + c;
  float ev[Nn];
  float mx = -1e30f;
#pragma unroll
  for (int j = 0; j < Nn; ++j) {
    float s = 1.f / (1.f + __expf(-ebase[(size_t)j << 9]));
    ev[j] = s;
    mx = fmaxf(mx, s);
  }
  float sum = 0.f;
#pragma unroll
  for (int j = 0; j < Nn; ++j) { float p = __expf(ev[j] - mx); ev[j] = p; sum += p; }
  const float* vb = xv + (((size_t)b * Nn) << 9) + c;
  float agg = 0.f;
#pragma unroll
  for (int j = 0; j < Nn; ++j) agg += ev[j] * vb[(size_t)j << 9];
  xtmp[idx] = xu[idx] + (agg / sum) * (1.f / (float)Nn);
}

// ---- node residual (float4): xOut = xtmp + relu(bn(xtmp)) ----------------
__global__ __launch_bounds__(256) void k_gnn_x_update(const float4* __restrict__ xtmp,
                                                      const float* __restrict__ mean,
                                                      const float* __restrict__ var,
                                                      const float* __restrict__ gw,
                                                      const float* __restrict__ gb,
                                                      float4* __restrict__ xOut) {
  int idx = blockIdx.x * 256 + threadIdx.x;
  if (idx >= XSZ / 4) return;
  int ch = (idx >> 7) % Nn;
  float scale = rsqrtf(var[ch] + BN_EPS) * gw[ch];
  float mu = mean[ch], bias = gb[ch];
  float4 v = xtmp[idx];
  float4 o;
  o.x = v.x + fmaxf((v.x - mu) * scale + bias, 0.f);
  o.y = v.y + fmaxf((v.y - mu) * scale + bias, 0.f);
  o.z = v.z + fmaxf((v.z - mu) * scale + bias, 0.f);
  o.w = v.w + fmaxf((v.w - mu) * scale + bias, 0.f);
  xOut[idx] = o;
}

extern "C" void kernel_launch(void* const* d_in, const int* in_sizes, int n_in,
                              void* d_out, int out_size, void* d_ws, size_t ws_size,
                              hipStream_t stream) {
  (void)in_sizes; (void)n_in; (void)out_size; (void)ws_size;
  const float* x_in  = (const float*)d_in[0];
  const float* e_in  = (const float*)d_in[1];
  const float* Wu    = (const float*)d_in[2];
  const float* Wv    = (const float*)d_in[3];
  const float* Wa    = (const float*)d_in[4];
  const float* Wb    = (const float*)d_in[5];
  const float* We    = (const float*)d_in[6];
  const float* bnv_w = (const float*)d_in[7];
  const float* bnv_b = (const float*)d_in[8];
  const float* bne_w = (const float*)d_in[9];
  const float* bne_b = (const float*)d_in[10];

  float* xO = (float*)d_out;        // [B, N, C]
  float* eO = xO + (size_t)XSZ;     // [B, N*N, C]

  // workspace layout (floats): xu | xv | xa | xb | m | stats(2048) | bf16 planes
  float* f    = (float*)d_ws;
  float* xu_b = f;
  float* xv_b = f + (size_t)XSZ;
  float* xa_b = f + 2 * (size_t)XSZ;   // reused as xtmp after edge update
  float* xb_b = f + 3 * (size_t)XSZ;
  float* m_b  = f + 4 * (size_t)XSZ;
  float* st   = m_b + (size_t)ESZ;
  float* mean_e = st;
  float* var_e  = st + NNE;
  float* mean_v = st + 2 * NNE;
  float* var_v  = st + 2 * NNE + Nn;
  __bf16* whi = (__bf16*)(st + 2048);       // 5 matrices, hi planes
  __bf16* wlo = whi + 5 * (size_t)CC2;      // 5 matrices, lo planes

  for (int l = 0; l < 2; ++l) {
    const float* Wm[5] = {Wu + (size_t)l * CC2, Wv + (size_t)l * CC2, Wa + (size_t)l * CC2,
                          Wb + (size_t)l * CC2, We + (size_t)l * CC2};
    for (int mi = 0; mi < 5; ++mi)
      k_gnn_wsplit<<<CC2 / 256, 256, 0, stream>>>(Wm[mi], whi + (size_t)mi * CC2,
                                                  wlo + (size_t)mi * CC2, CC2);

    const float* xcur = l ? (const float*)xO : x_in;
    const float* ecur = l ? (const float*)eO : e_in;

    // 4 * 108 * 8 waves * 32 lanes / 256 = 432 blocks
    k_gnn_gemm_x<<<(4 * (BN_ROWS / 16) * NTG * 32) / 256, 256, 0, stream>>>(
        xcur, whi, wlo, xu_b, xv_b, xa_b, xb_b);
    // 2916 * 8 waves * 32 lanes / 256 = 2916 blocks
    k_gnn_gemm_edge<<<((BE_ROWS / 16) * NTG * 32) / 256, 256, 0, stream>>>(
        ecur, whi + 4 * (size_t)CC2, wlo + 4 * (size_t)CC2, xa_b, xb_b, m_b);

    k_gnn_bnstats<<<NNE, 256, 0, stream>>>(m_b, NNE, mean_e, var_e);
    k_gnn_edge_update<<<(ESZ / 4) / 256, 256, 0, stream>>>(
        (const float4*)ecur, (const float4*)m_b, mean_e, var_e,
        bne_w + (size_t)l * NNE, bne_b + (size_t)l * NNE, (float4*)eO);

    k_gnn_softmax_agg<<<XSZ / 256, 256, 0, stream>>>(eO, xv_b, xu_b, xa_b);
    k_gnn_bnstats<<<Nn, 256, 0, stream>>>(xa_b, Nn, mean_v, var_v);
    k_gnn_x_update<<<(XSZ / 4) / 256, 256, 0, stream>>>(
        (const float4*)xa_b, mean_v, var_v,
        bnv_w + (size_t)l * Nn, bnv_b + (size_t)l * Nn, (float4*)xO);
  }
}
